// MultiHeadAttentionMechanism_44573170598626
// MI455X (gfx1250) — matrix operand order repaired and checked by reading, not compiled, not run on hardware
//
#include <hip/hip_runtime.h>
#include <hip/hip_bf16.h>

// ---------------------------------------------------------------------------
// GAT-style dense attention, single pass over 512MB `edges` (HBM-bound ~22us).
//   num[n,h,d] = sum_m exp(tanh(edges[n,m,:].W[:,h])) * feat[m,d]
//   den[n,h]   = sum_m exp(tanh(...))
//   out[n,d]   = tanh( mean_h num/den )
// Both GEMMs run on v_wmma_f32_16x16x32_f16. The projection D-tile layout
// coincides (per CDNA5 ISA fragment tables) with the A-fragment layout of the
// accumulation GEMM, so attention weights never leave their lanes.
// ---------------------------------------------------------------------------

#define N_NODES 2048
#define D_EDGE  32
#define D_FEAT  128
#define NHEADS  4
#define SUPER   512            // m super-chunk staged in LDS
#define LDST    (SUPER + 8)    // padded f16 row stride (1040B: 16B-aligned rows,
                               // 4-bank rotation per d -> conflict-free b128 reads)
#define WAVES   8
#define BLOCK   (WAVES * 32)

typedef __attribute__((ext_vector_type(16))) _Float16 v16h;
typedef __attribute__((ext_vector_type(8)))  _Float16 v8h;
typedef __attribute__((ext_vector_type(8)))  float    v8f;

__device__ __forceinline__ float fexp(float x) {
  return __builtin_amdgcn_exp2f(x * 1.4426950408889634f);
}
__device__ __forceinline__ float ftanh(float x) {
  // tanh(x) = 1 - 2/(1 + e^{2x}); saturates correctly at +-inf.
  float e2 = __builtin_amdgcn_exp2f(x * 2.8853900817779268f);
  return 1.0f - 2.0f * __builtin_amdgcn_rcpf(1.0f + e2);
}

__global__ __launch_bounds__(BLOCK) void gat_attn_kernel(
    const float* __restrict__ edges,     // [N, N, 32]
    const float* __restrict__ features,  // [N, 128]
    const float* __restrict__ W,         // [32, 4]
    float* __restrict__ out)             // [N, 128]
{
  extern __shared__ _Float16 ldsT[];     // featT[d][m_local], [128][LDST] f16

  const int tid  = threadIdx.x;
  const int wave = tid >> 5;
  const int lane = tid & 31;
  const int hlf  = lane >> 4;            // wave half (K-group selector)
  const int col  = lane & 15;            // fragment row/col index
  const int n    = blockIdx.x * WAVES + wave;   // node owned by this wave

  // Constant projection B-fragment: W zero-padded to [32 x 16].
  // B layout: elems 0-7: K=8*hlf+0..7, elems 8-15: K=16+8*hlf+0..7 (matches A).
  v16h bW;
#pragma unroll
  for (int j = 0; j < 16; ++j) {
    int e = (j < 8) ? (8 * hlf + j) : (16 + 8 * hlf + (j - 8));
    bW[j] = (col < NHEADS) ? (_Float16)W[e * NHEADS + col] : (_Float16)0.0f;
  }

  v8f acc[8];                            // [16 rows (h padded) x 128 d] accum
#pragma unroll
  for (int t = 0; t < 8; ++t) acc[t] = {};
  float denom = 0.0f;                    // partial for head = col (col<4 valid)

  const float* erow = edges + (size_t)n * N_NODES * D_EDGE;

  for (int ms = 0; ms < N_NODES; ms += SUPER) {
    __syncthreads();                     // prior super-chunk consumers done
    // ---- stage features[ms:ms+SUPER][:] transposed f32->f16 into LDS ------
    for (int base = 0; base < SUPER * (D_FEAT / 4); base += BLOCK) {
      int idx = base + tid;
      int mi      = idx & 7;             // 8 consecutive m per dq group
      int r       = idx >> 3;
      int dq      = r & 31;              // 32 float4 quads cover d=0..127
      int m_local = ((r >> 5) << 3) + mi;
      const float4 f =
          *(const float4*)(features + (size_t)(ms + m_local) * D_FEAT + dq * 4);
      int d0 = dq * 4;
      ldsT[(d0 + 0) * LDST + m_local] = (_Float16)f.x;
      ldsT[(d0 + 1) * LDST + m_local] = (_Float16)f.y;
      ldsT[(d0 + 2) * LDST + m_local] = (_Float16)f.z;
      ldsT[(d0 + 3) * LDST + m_local] = (_Float16)f.w;
    }
    __syncthreads();

    // ---- main loop: 32 neighbors per iteration -----------------------------
    for (int mc = 0; mc < SUPER; mc += 32) {
      v16h aAtt;                         // A-frag [16(h) x 32(m)] of exp(tanh)
#pragma unroll
      for (int t = 0; t < 2; ++t) {      // two 16-m projection tiles
        // A-frag (16-bit 16x32): this lane = m-row (col), K=e in groups
        //   regs 0-3: e = 8*hlf+0..7,  regs 4-7: e = 16+8*hlf+0..7
        const float* rp =
            erow + (size_t)(ms + mc + t * 16 + col) * D_EDGE + 8 * hlf;
        float4 e0 = *(const float4*)(rp + 0);
        float4 e1 = *(const float4*)(rp + 4);
        float4 e2 = *(const float4*)(rp + 16);
        float4 e3 = *(const float4*)(rp + 20);
        v16h aP;
        aP[0]  = (_Float16)e0.x; aP[1]  = (_Float16)e0.y;
        aP[2]  = (_Float16)e0.z; aP[3]  = (_Float16)e0.w;
        aP[4]  = (_Float16)e1.x; aP[5]  = (_Float16)e1.y;
        aP[6]  = (_Float16)e1.z; aP[7]  = (_Float16)e1.w;
        aP[8]  = (_Float16)e2.x; aP[9]  = (_Float16)e2.y;
        aP[10] = (_Float16)e2.z; aP[11] = (_Float16)e2.w;
        aP[12] = (_Float16)e3.x; aP[13] = (_Float16)e3.y;
        aP[14] = (_Float16)e3.z; aP[15] = (_Float16)e3.w;

        v8f z = {};
        v8f dp = __builtin_amdgcn_wmma_f32_16x16x32_f16(
            false, aP, false, bW, (short)0, z, false, false);

        // D layout: this lane = column h (=col), reg j = m-row j + 8*hlf.
        // That is exactly element t*8+j of the accumulation A-fragment.
#pragma unroll
        for (int j = 0; j < 8; ++j) {
          float w = fexp(ftanh(dp[j]));  // padded cols: exp(tanh(0)) = 1, benign
          aAtt[t * 8 + j] = (_Float16)w;
          denom += w;
        }
      }

      // speculative prefetch of next edge chunk (global_prefetch_b8)
      __builtin_prefetch(erow + (size_t)(ms + mc + 32 + col) * D_EDGE, 0, 0);

      // ---- rank-32 update: acc[h,d] += aAtt[h,m] * featT[d,m] -------------
#pragma unroll
      for (int dt = 0; dt < 8; ++dt) {
        int d = dt * 16 + col;
        // B-frag: this lane = column d, K = mc + 8*hlf + 0..7, mc + 16 + 8*hlf + 0..7
        const v8h* bp = (const v8h*)(ldsT + (size_t)d * LDST + mc + 8 * hlf);
        union { v16h v; v8h h[2]; } bu;
        bu.h[0] = bp[0];                 // ds_load_b128
        bu.h[1] = bp[2];                 // ds_load_b128
        acc[dt] = __builtin_amdgcn_wmma_f32_16x16x32_f16(
            false, aAtt, false, bu.v, (short)0, acc[dt], false, false);
      }
    }
  }

  // ---- epilogue: combine halves, broadcast denominators, tanh(mean) -------
  denom += __shfl_xor(denom, 16, 32);    // lanes h and h+16 now both hold total
  float r[4];
#pragma unroll
  for (int h = 0; h < 4; ++h)
    r[h] = __builtin_amdgcn_rcpf(__shfl(denom, h, 32));

  if (hlf == 0) {                        // regs 0-3 hold rows h=0..3
#pragma unroll
    for (int dt = 0; dt < 8; ++dt) {
      float s = acc[dt][0] * r[0] + acc[dt][1] * r[1] +
                acc[dt][2] * r[2] + acc[dt][3] * r[3];
      out[(size_t)n * D_FEAT + dt * 16 + col] = ftanh(0.25f * s);
    }
  }
}

extern "C" void kernel_launch(void* const* d_in, const int* in_sizes, int n_in,
                              void* d_out, int out_size, void* d_ws, size_t ws_size,
                              hipStream_t stream) {
  (void)in_sizes; (void)n_in; (void)out_size; (void)d_ws; (void)ws_size;
  const float* edges    = (const float*)d_in[0];
  const float* features = (const float*)d_in[1];
  const float* W        = (const float*)d_in[2];
  float*       out      = (float*)d_out;

  const size_t shmem = (size_t)D_FEAT * LDST * sizeof(_Float16); // 130 KB < 320 KB
  // CDNA5 allows up to 320KB LDS per workgroup; raise the dynamic-LDS cap.
  hipFuncSetAttribute((const void*)gat_attn_kernel,
                      hipFuncAttributeMaxDynamicSharedMemorySize, (int)shmem);

  gat_attn_kernel<<<N_NODES / WAVES, BLOCK, shmem, stream>>>(
      edges, features, W, out);
}
